// DSALayer_68401649156444
// MI455X (gfx1250) — matrix operand order repaired and checked by reading, not compiled, not run on hardware
//
#include <hip/hip_runtime.h>
#include <hip/hip_bf16.h>
#include <math.h>

// ---------------- problem constants (match reference) ----------------
#define B_    4
#define T_    1024
#define D_    1024
#define H_    16
#define HD_   64          // head dim
#define DN    (3 * D_)    // 3072
#define TOPK_ 32
#define DECAY_     0.9f
#define THRESH_    1.0f
#define LN_EPS_    1e-5f

typedef _Float16 h16 __attribute__((ext_vector_type(16)));
typedef _Float16 h8  __attribute__((ext_vector_type(8)));
typedef float    f8  __attribute__((ext_vector_type(8)));
typedef float    f4  __attribute__((ext_vector_type(4)));

// D = A(16x32 f16) * B(32x16 f16) + C(16x16 f32)  -- one tile per wave32
__device__ __forceinline__ f8 wmma_f32_16x16x32(h16 a, h16 b, f8 c) {
    return __builtin_amdgcn_wmma_f32_16x16x32_f16(
        /*neg_a=*/false, a, /*neg_b=*/false, b,
        /*c_mod=*/(short)0, c, /*reuse_a=*/false, /*reuse_b=*/false);
}

__device__ __forceinline__ h16 load_a_tile(const _Float16* ap) {
    // A-layout: lane holds row (lane&15); halfs = base + {0..15}, base = 16*(lane>>4)
    h8 x0 = *(const h8*)ap;
    h8 x1 = *(const h8*)(ap + 8);
    h16 av;
#pragma unroll
    for (int i = 0; i < 8; ++i) { av[i] = x0[i]; av[i + 8] = x1[i]; }
    return av;
}

// ---------------- small utility kernels ----------------
__global__ void k_init_acc(float* acc) {
    if (threadIdx.x < 2) acc[threadIdx.x] = 0.0f;
}

__global__ void k_abssum(const float* __restrict__ w, int n, float* __restrict__ acc) {
    __shared__ float sm[256];
    float s = 0.0f;
    for (int i = blockIdx.x * blockDim.x + threadIdx.x; i < n; i += gridDim.x * blockDim.x)
        s += fabsf(w[i]);
    sm[threadIdx.x] = s;
    __syncthreads();
    for (int o = 128; o > 0; o >>= 1) {
        if (threadIdx.x < o) sm[threadIdx.x] += sm[threadIdx.x + o];
        __syncthreads();
    }
    if (threadIdx.x == 0) atomicAdd(acc, sm[0]);
}

// BitNet b1.58 ternary quantization -> f16 weights (values in {-s,0,s})
__global__ void k_quant_f16(const float* __restrict__ w, int n,
                            const float* __restrict__ acc, float inv_n,
                            _Float16* __restrict__ out) {
    int i = blockIdx.x * blockDim.x + threadIdx.x;
    if (i >= n) return;
    float scale = (*acc) * inv_n + 1e-8f;
    float q = rintf(w[i] / scale);
    q = fminf(1.0f, fmaxf(-1.0f, q));
    out[i] = (_Float16)(q * scale);
}

__global__ void k_f32_to_f16(const float* __restrict__ x, int n, _Float16* __restrict__ y) {
    int i = blockIdx.x * blockDim.x + threadIdx.x;
    if (i < n) y[i] = (_Float16)x[i];
}

// ---------------- WMMA GEMM:  out[M,N] = A[M,K] @ W[N,K]^T + bias ----------------
// Each wave computes a 16x64 output strip: one A-load feeds 4 WMMAs.
// OUTMODE 0: store f16; OUTMODE 1: store f32
template <int OUTMODE>
__global__ void k_gemm_bitlinear(const _Float16* __restrict__ A,
                                 const _Float16* __restrict__ W,
                                 const float* __restrict__ bias,
                                 void* __restrict__ out,
                                 int M, int N, int K) {
    // wave id as SGPR: scalar branches + scalar addressing
    int wave = __builtin_amdgcn_readfirstlane((blockIdx.x * blockDim.x + threadIdx.x) >> 5);
    int lane = threadIdx.x & 31;
    int ngrp = N >> 6;                         // 64-wide column groups
    int tm  = wave / ngrp;
    int tg  = wave % ngrp;
    if (tm >= (M >> 4)) return;                // scalar-uniform

    int l15  = lane & 15;
    int ksel = lane >> 4;
    int mrow = tm * 16 + l15;
    int nbase = tg * 64 + l15;                 // column for B sub-tile 0

    f8 c0 = {}, c1 = {}, c2 = {}, c3 = {};
    for (int kt = 0; kt < K; kt += 32) {
        h16 av = load_a_tile(A + (long)mrow * K + kt + ksel * 16);
        const _Float16* bp = W + (long)nbase * K + kt + ksel * 16;
        h16 b0 = *(const h16*)bp;
        h16 b1 = *(const h16*)(bp + 16L * K);
        h16 b2 = *(const h16*)(bp + 32L * K);
        h16 b3 = *(const h16*)(bp + 48L * K);
        // hint next k-step's weight strips into cache (global_prefetch_b8)
        if (kt + 32 < K) {
            __builtin_prefetch(bp + 32, 0, 2);
            __builtin_prefetch(bp + 16L * K + 32, 0, 2);
            __builtin_prefetch(bp + 32L * K + 32, 0, 2);
            __builtin_prefetch(bp + 48L * K + 32, 0, 2);
        }
        c0 = wmma_f32_16x16x32(av, b0, c0);
        c1 = wmma_f32_16x16x32(av, b1, c1);
        c2 = wmma_f32_16x16x32(av, b2, c2);
        c3 = wmma_f32_16x16x32(av, b3, c3);
    }
#pragma unroll
    for (int r = 0; r < 8; ++r) {
        int m = tm * 16 + r + ksel * 8;
#pragma unroll
        for (int t = 0; t < 4; ++t) {
            int n = nbase + t * 16;
            float v = (t == 0 ? c0[r] : t == 1 ? c1[r] : t == 2 ? c2[r] : c3[r]) + bias[n];
            if (OUTMODE == 0) ((_Float16*)out)[(long)m * N + n] = (_Float16)v;
            else              ((float*)out)[(long)m * N + n] = v;
        }
    }
}

// ---------------- transpose V -> vT[b,h,d,T] f16 (contiguous B-operand for P@V) ----------------
__global__ void k_vT(const _Float16* __restrict__ qkv, _Float16* __restrict__ vT) {
    long i = (long)blockIdx.x * blockDim.x + threadIdx.x;
    if (i >= (long)B_ * H_ * HD_ * T_) return;
    int t = (int)(i % T_);  long r2 = i / T_;
    int d = (int)(r2 % HD_); long r3 = r2 / HD_;
    int h = (int)(r3 % H_);  int b = (int)(r3 / H_);
    vT[i] = qkv[(long)(b * T_ + t) * DN + 2 * D_ + h * HD_ + d];
}

// ---------------- fused attention: scores -> top-k -> softmax -> probs + P@V ----------------
// One block (8 waves) per (b, h, 16-row q-tile). Scores live in LDS; only the
// mandated attn_probs write touches HBM for the TxT matrix, and that write is
// issued with GLOBAL_STORE_ASYNC_FROM_LDS_B128 so it overlaps the P@V WMMAs.
#define SSTR 1028   // padded LDS row stride (floats); 1028 % 64 == 4 -> conflict-free A reads
__global__ void k_attn_fused(const _Float16* __restrict__ qkv,
                             const _Float16* __restrict__ vT,
                             float* __restrict__ attn,
                             _Float16* __restrict__ ctx) {
    __shared__ float sv[16 * SSTR];            // ~64.25 KB score slab
    __shared__ float rv[256];
    __shared__ int   ri[256];
    __shared__ float topv[16][TOPK_];
    __shared__ int   topi[16][TOPK_];
    __shared__ float sden[16];

    int tid  = threadIdx.x;
    int w    = __builtin_amdgcn_readfirstlane(tid >> 5);  // SGPR wave id
    int lane = tid & 31;
    int l15  = lane & 15;
    int ksel = lane >> 4;

    int bh = blockIdx.x / (T_ / 16);
    int qt = blockIdx.x % (T_ / 16);
    int b = bh / H_, h = bh % H_;

    // ---- phase A: scores into LDS (wave w owns k-tiles {w, w+8, ..., w+56}) ----
    int qrow = qt * 16 + l15;
    const _Float16* qbase = qkv + (long)(b * T_ + qrow) * DN + h * HD_;
    h16 aq0 = load_a_tile(qbase + 0 + ksel * 16);   // d = 0..31
    h16 aq1 = load_a_tile(qbase + 32 + ksel * 16);  // d = 32..63

    for (int j = 0; j < 8; ++j) {
        int kt2 = w + 8 * j;                   // SGPR -> scalar branch below
        int kcol = kt2 * 16 + l15;
        if (kt2 <= qt) {                       // s_cbranch, EXEC stays all-ones
            const _Float16* bp = qkv + (long)(b * T_ + kcol) * DN + D_ + h * HD_;
            h16 b0 = *(const h16*)(bp + ksel * 16);
            h16 b1 = *(const h16*)(bp + 32 + ksel * 16);
            f8 c = {};
            c = wmma_f32_16x16x32(aq0, b0, c);
            c = wmma_f32_16x16x32(aq1, b1, c);
#pragma unroll
            for (int r = 0; r < 8; ++r) {
                int m = r + ksel * 8;
                float v = (kcol <= qt * 16 + m) ? c[r] * 0.125f : -INFINITY;
                sv[m * SSTR + kcol] = v;
            }
        } else {
#pragma unroll
            for (int r = 0; r < 8; ++r)
                sv[(r + ksel * 8) * SSTR + kcol] = -INFINITY;
        }
    }
    __syncthreads();

    // ---- phase B: top-32 per row (16 threads per row, repeated argmax) ----
    int row = tid >> 4;                        // 0..15
    int s   = tid & 15;
    float* srow = sv + row * SSTR;
    for (int it = 0; it < TOPK_; ++it) {
        float bvv = -INFINITY; int bii = 0x7fffffff;
        for (int i = s; i < T_; i += 16) {
            float v = srow[i];
            if (v > bvv || (v == bvv && i < bii)) { bvv = v; bii = i; }
        }
        rv[tid] = bvv; ri[tid] = bii;
        __syncthreads();
        for (int o = 8; o > 0; o >>= 1) {
            if (s < o) {
                float v2 = rv[tid + o]; int i2 = ri[tid + o];
                if (v2 > rv[tid] || (v2 == rv[tid] && i2 < ri[tid])) { rv[tid] = v2; ri[tid] = i2; }
            }
            __syncthreads();
        }
        if (s == 0) {
            topv[row][it] = rv[tid]; topi[row][it] = ri[tid];
            if (ri[tid] < T_) srow[ri[tid]] = -INFINITY;
        }
        __syncthreads();
    }
    if (tid < 16) {                            // per-row softmax denominator
        float m = topv[tid][0];                // iteration 0 found the row max
        float ssum = 0.0f;
        for (int j = 0; j < TOPK_; ++j) ssum += expf(topv[tid][j] - m);  // exp(-inf)=0
        sden[tid] = ssum;
    }
    __syncthreads();

    // ---- rebuild probs in LDS: zero slab, scatter 32 per row ----
    for (int i = tid; i < 16 * SSTR; i += 256) sv[i] = 0.0f;
    __syncthreads();
    for (int t2 = tid; t2 < 16 * TOPK_; t2 += 256) {
        int r2 = t2 >> 5, j2 = t2 & 31;
        float e = expf(topv[r2][j2] - topv[r2][0]) / sden[r2];
        sv[r2 * SSTR + topi[r2][j2]] = e;      // dup -inf indices write 0: benign
    }
    __syncthreads();

    // ---- mandated attn_probs write: async LDS -> global, overlaps phase C ----
    // 16 rows x 1024 floats = 4096 x 16B chunks; 16 chunks per thread.
    {
        float* orow = attn + ((long)bh * T_ + qt * 16) * T_;
#pragma unroll
        for (int j = 0; j < 16; ++j) {
            int cidx = tid + 256 * j;          // chunk index
            int r2 = cidx >> 8;                // row 0..15
            int c2 = (cidx & 255) * 4;         // float column
            // LDS aperture rule: generic addr [31:0] == LDS byte offset
            unsigned lds_off = (unsigned)(unsigned long long)(const void*)(sv + r2 * SSTR + c2);
            unsigned long long gaddr = (unsigned long long)(const void*)(orow + (long)r2 * T_ + c2);
            asm volatile("global_store_async_from_lds_b128 %0, %1, off"
                         :: "v"(gaddr), "v"(lds_off) : "memory");
        }
    }
    // S_ENDPGM performs an implicit wait-idle, covering ASYNCcnt completion.

    // ---- phase C: ctx(16x64) = P @ V straight from LDS (waves 0..3, scalar branch) ----
    if (w < 4) {
        int dcol = w * 16 + l15;
        const _Float16* vrow = vT + ((long)bh * HD_ + dcol) * T_;
        f8 c = {};
        int klim = (((qt + 1) * 16) + 31) & ~31;   // causal K limit
        for (int kt = 0; kt < klim; kt += 32) {
            const float* ap = sv + l15 * SSTR + kt + ksel * 16;
            f4 p0 = *(const f4*)(ap);
            f4 p1 = *(const f4*)(ap + 4);
            f4 p2 = *(const f4*)(ap + 8);
            f4 p3 = *(const f4*)(ap + 12);
            h16 av;
#pragma unroll
            for (int i = 0; i < 4; ++i) {
                av[i]      = (_Float16)p0[i];
                av[i + 4]  = (_Float16)p1[i];
                av[i + 8]  = (_Float16)p2[i];
                av[i + 12] = (_Float16)p3[i];
            }
            h16 bv = *(const h16*)(vrow + kt + ksel * 16);
            c = wmma_f32_16x16x32(av, bv, c);
        }
#pragma unroll
        for (int r = 0; r < 8; ++r) {
            int m = qt * 16 + r + ksel * 8;
            ctx[(long)(b * T_ + m) * D_ + h * HD_ + dcol] = (_Float16)c[r];
        }
    }
}

// ---------------- layer norm per row of D ----------------
__global__ void k_ln(float* __restrict__ x_all, const float* __restrict__ gamma,
                     const float* __restrict__ beta) {
    float* x = x_all + (long)blockIdx.x * D_;
    __shared__ float s1[256], s2[256];
    int tid = threadIdx.x;
    float a = 0.0f, b = 0.0f;
    for (int i = tid; i < D_; i += 256) { float v = x[i]; a += v; b += v * v; }
    s1[tid] = a; s2[tid] = b;
    __syncthreads();
    for (int o = 128; o > 0; o >>= 1) {
        if (tid < o) { s1[tid] += s1[tid + o]; s2[tid] += s2[tid + o]; }
        __syncthreads();
    }
    float mu  = s1[0] * (1.0f / D_);
    float var = s2[0] * (1.0f / D_) - mu * mu;
    float inv = rsqrtf(var + LN_EPS_);
    for (int i = tid; i < D_; i += 256) x[i] = (x[i] - mu) * inv * gamma[i] + beta[i];
}

// ---------------- LIF neuron scan over T (thread per (b,d), coalesced in d) ----------------
__global__ void k_lif(const float* __restrict__ analog, float* __restrict__ out) {
    int i = blockIdx.x * blockDim.x + threadIdx.x;
    if (i >= B_ * D_) return;
    int b = i / D_, d = i % D_;
    float mem = 0.0f;
    for (int t = 0; t < T_; ++t) {
        float a = analog[(long)(b * T_ + t) * D_ + d];
        mem = mem * DECAY_ + a;
        float s = (mem > THRESH_) ? 1.0f : 0.0f;
        mem -= s * THRESH_;
        out[(long)(b * T_ + t) * D_ + d] = s;
    }
}

// ---------------- host orchestration ----------------
extern "C" void kernel_launch(void* const* d_in, const int* in_sizes, int n_in,
                              void* d_out, int out_size, void* d_ws, size_t ws_size,
                              hipStream_t stream) {
    (void)in_sizes; (void)n_in; (void)out_size; (void)ws_size;
    const float* x      = (const float*)d_in[0];
    const float* w_qkv  = (const float*)d_in[1];
    const float* b_qkv  = (const float*)d_in[2];
    const float* w_out  = (const float*)d_in[3];
    const float* b_out  = (const float*)d_in[4];
    const float* ln_g   = (const float*)d_in[5];
    const float* ln_b   = (const float*)d_in[6];

    float* out_spikes = (float*)d_out;                          // [B,T,D]
    float* attn       = (float*)d_out + (size_t)B_ * T_ * D_;   // [B,H,T,T]

    char*  ws  = (char*)d_ws;
    size_t off = 0;
    auto alloc = [&](size_t bytes) -> void* {
        void* p = ws + off; off += (bytes + 255) & ~(size_t)255; return p;
    };
    float*     acc     = (float*)alloc(256);
    _Float16*  xh      = (_Float16*)alloc((size_t)B_ * T_ * D_ * 2);      //  8 MB
    _Float16*  wqkv_q  = (_Float16*)alloc((size_t)3 * D_ * D_ * 2);       //  6 MB
    _Float16*  wout_q  = (_Float16*)alloc((size_t)D_ * D_ * 2);           //  2 MB
    _Float16*  qkv     = (_Float16*)alloc((size_t)B_ * T_ * 3 * D_ * 2);  // 24 MB
    _Float16*  vT      = (_Float16*)alloc((size_t)B_ * H_ * HD_ * T_ * 2);//  8 MB
    _Float16*  ctx     = (_Float16*)alloc((size_t)B_ * T_ * D_ * 2);      //  8 MB
    float*     analog  = (float*)alloc((size_t)B_ * T_ * D_ * 4);         // 16 MB

    // 1) BitNet weight quantization
    k_init_acc<<<1, 32, 0, stream>>>(acc);
    k_abssum<<<512, 256, 0, stream>>>(w_qkv, 3 * D_ * D_, acc + 0);
    k_abssum<<<512, 256, 0, stream>>>(w_out, D_ * D_, acc + 1);
    k_quant_f16<<<(3 * D_ * D_ + 255) / 256, 256, 0, stream>>>(w_qkv, 3 * D_ * D_, acc + 0,
                                                               1.0f / (3 * D_ * D_), wqkv_q);
    k_quant_f16<<<(D_ * D_ + 255) / 256, 256, 0, stream>>>(w_out, D_ * D_, acc + 1,
                                                           1.0f / (D_ * D_), wout_q);
    k_f32_to_f16<<<(B_ * T_ * D_ + 255) / 256, 256, 0, stream>>>(x, B_ * T_ * D_, xh);

    // 2) QKV projection: [4096,3072] = xh @ Wq^T + b   (waves = 256 * 48)
    k_gemm_bitlinear<0><<<(256 * 48) / 8, 256, 0, stream>>>(xh, wqkv_q, b_qkv, qkv,
                                                            B_ * T_, 3 * D_, D_);

    // 3) V transpose, then fused scores/top-k/softmax/P@V (block per (b,h,q-tile))
    k_vT<<<(B_ * H_ * HD_ * T_ + 255) / 256, 256, 0, stream>>>(qkv, vT);
    k_attn_fused<<<B_ * H_ * (T_ / 16), 256, 0, stream>>>(qkv, vT, attn, ctx);

    // 4) output projection: [4096,1024] = ctx @ Wout^T + b  (waves = 256 * 16)
    k_gemm_bitlinear<1><<<(256 * 16) / 8, 256, 0, stream>>>(ctx, wout_q, b_out, analog,
                                                            B_ * T_, D_, D_);

    // 5) layer norm + LIF scan
    k_ln<<<B_ * T_, 256, 0, stream>>>(analog, ln_g, ln_b);
    k_lif<<<(B_ * D_ + 255) / 256, 256, 0, stream>>>(analog, out_spikes);
}
